// MultiAspectRetrieval_21715354649928
// MI455X (gfx1250) — compile-verified, matrix operand order checked by reading, not run on hardware
//
#include <hip/hip_runtime.h>
#include <hip/hip_bf16.h>
#include <math.h>

// ---------------- problem constants (from reference) ----------------
#define S_ASP   4
#define C_CLU   1024
#define N_KEYS  131072
#define DKK     128
#define DAA     2048
#define M_TOP   32
#define K_MAX_  64
#define B_B     64
#define NPC     (N_KEYS / C_CLU)     // 128 keys per cluster
#define MCAND   (M_TOP * NPC)        // 4096 candidates
#define EPS_    1e-8f
#define NEG_INF (-3.0e38f)

typedef __attribute__((ext_vector_type(16))) __bf16 v16bf;
typedef __attribute__((ext_vector_type(8)))  __bf16 v8bf;
typedef __attribute__((ext_vector_type(8)))  float  v8f;

// ---------------- workspace layout (byte offsets, 256B aligned) ------
#define WS_W      0          // float[4]           softmax(aspect_weights)
#define WS_QNF    256        // float[4][64][128]  normalized queries f32
#define WS_QNBF   131328     // __bf16[4][64][128] normalized queries bf16
#define WS_CSCORE 196864     // float[64][1024]    centroid scores
#define WS_TOPC   459008     // int[64][32]        top clusters
#define WS_RMAX   467200     // float[64]          row max of s_full
#define WS_RSUM   467456     // float[64]          row sum-exp of s_full

__device__ __forceinline__ float sigmoidf_(float x) { return 1.0f / (1.0f + expf(-x)); }

// ===================== K0: w = softmax(aspect_weights) ===============
__global__ void k0_prep_w(const float* __restrict__ aw, float* __restrict__ w) {
    if (threadIdx.x == 0) {
        float m = aw[0];
        for (int s = 1; s < S_ASP; ++s) m = fmaxf(m, aw[s]);
        float e[S_ASP], sum = 0.f;
        for (int s = 0; s < S_ASP; ++s) { e[s] = expf(aw[s] - m); sum += e[s]; }
        for (int s = 0; s < S_ASP; ++s) w[s] = e[s] / sum;
    }
}

// ============ K1: queries = W_Q @ z, L2-normalize, store f32 + bf16 ==
// grid: B*S blocks of 128 threads; block = b*S + s; thread = k
__global__ __launch_bounds__(128) void k1_queries(
        const float* __restrict__ z, const float* __restrict__ WQ,
        float* __restrict__ qnf, __bf16* __restrict__ qnbf) {
    __shared__ float zsh[DAA];
    __shared__ float red[128];
    const int b = blockIdx.x / S_ASP;
    const int s = blockIdx.x % S_ASP;
    const int k = threadIdx.x;
    for (int i = k; i < DAA; i += 128) zsh[i] = z[(size_t)b * DAA + i];
    __syncthreads();
    const float* wrow = WQ + ((size_t)s * DKK + k) * DAA;
    float q = 0.f;
    for (int a = 0; a < DAA; ++a) q = fmaf(wrow[a], zsh[a], q);
    red[k] = q * q;
    __syncthreads();
    for (int off = 64; off > 0; off >>= 1) {
        if (k < off) red[k] += red[k + off];
        __syncthreads();
    }
    const float rn = 1.0f / (sqrtf(red[0]) + EPS_);
    const float qn = q * rn;
    const size_t idx = ((size_t)s * B_B + b) * DKK + k;
    qnf[idx]  = qn;
    qnbf[idx] = (__bf16)qn;
}

// ======= K2: c_score[b,c] = sum_s w[s] * qn[b,s]·(cent/|cent|) =======
// grid: B blocks of 256 threads
__global__ __launch_bounds__(256) void k2_centroids(
        const float* __restrict__ qnf, const float* __restrict__ cent,
        const float* __restrict__ w, float* __restrict__ cscore) {
    __shared__ float qsh[S_ASP][DKK];
    __shared__ float wsh[S_ASP];
    const int b = blockIdx.x, tid = threadIdx.x;
    for (int i = tid; i < S_ASP * DKK; i += 256) {
        int s = i / DKK, k = i % DKK;
        qsh[s][k] = qnf[((size_t)s * B_B + b) * DKK + k];
    }
    if (tid < S_ASP) wsh[tid] = w[tid];
    __syncthreads();
    for (int c = tid; c < C_CLU; c += 256) {
        float score = 0.f;
        for (int s = 0; s < S_ASP; ++s) {
            const float* cv = cent + ((size_t)s * C_CLU + c) * DKK;
            float dot = 0.f, ss = 0.f;
            for (int k = 0; k < DKK; ++k) {
                float v = cv[k];
                dot = fmaf(qsh[s][k], v, dot);
                ss  = fmaf(v, v, ss);
            }
            score += wsh[s] * dot / (sqrtf(ss) + EPS_);
        }
        cscore[(size_t)b * C_CLU + c] = score;
    }
}

// ================= K3: top-32 clusters per batch row =================
__global__ __launch_bounds__(256) void k3_topclusters(
        const float* __restrict__ cscore, int* __restrict__ topc) {
    __shared__ float sc[C_CLU];
    __shared__ float rv[256];
    __shared__ int   ri[256];
    const int b = blockIdx.x, tid = threadIdx.x;
    for (int i = tid; i < C_CLU; i += 256) sc[i] = cscore[(size_t)b * C_CLU + i];
    __syncthreads();
    for (int it = 0; it < M_TOP; ++it) {
        float bv = NEG_INF; int bi = 0;
        for (int i = tid; i < C_CLU; i += 256)
            if (sc[i] > bv) { bv = sc[i]; bi = i; }
        rv[tid] = bv; ri[tid] = bi;
        __syncthreads();
        for (int off = 128; off > 0; off >>= 1) {
            if (tid < off && rv[tid + off] > rv[tid]) { rv[tid] = rv[tid + off]; ri[tid] = ri[tid + off]; }
            __syncthreads();
        }
        if (tid == 0) { topc[b * M_TOP + it] = ri[0]; sc[ri[0]] = NEG_INF; }
        __syncthreads();
    }
}

// ============ K4: dense WMMA pass, s_full = sum_s w_s qn·k_norm ======
// grid: N/TN blocks of 256 threads (8 waves). Wave w: m-tile = w&3 (16 rows
// of B), group = w>>2 picks 4 of the 8 local n-tiles. bf16 WMMA, keys
// streamed f32->bf16 through LDS with fused sum-of-squares.
#define TN 128
__global__ __launch_bounds__(256) void k4_dense(
        const float* __restrict__ pool, const float* __restrict__ w,
        const __bf16* __restrict__ qnbf, float* __restrict__ sfull) {
    __shared__ __bf16 kls[TN][DKK];   // 32 KB key tile (bf16)
    __shared__ float  ssq[TN];
    __shared__ float  rno[TN];
    const int nbase = blockIdx.x * TN;
    const int tid   = threadIdx.x;
    const int lane  = tid & 31;
    const int wave  = tid >> 5;
    const int mtile = wave & 3;
    const int grp   = wave >> 2;
    const int mrow  = (lane < 16) ? lane : (lane - 16);

    v8f acc[4] = {v8f{}, v8f{}, v8f{}, v8f{}};

    for (int s = 0; s < S_ASP; ++s) {
        for (int i = tid; i < TN; i += 256) ssq[i] = 0.f;
        __syncthreads();
        const float* kp = pool + ((size_t)s * N_KEYS + nbase) * DKK;
        // prefetch next aspect's tile start (gfx1250 global_prefetch_b8)
        if (s + 1 < S_ASP)
            __builtin_prefetch(pool + ((size_t)(s + 1) * N_KEYS + nbase) * DKK + tid * 16, 0, 1);
        // stream keys: f32 -> bf16 into LDS, fused sum of squares
        for (int idx = tid; idx < TN * (DKK / 4); idx += 256) {
            const int nl = idx >> 5;           // DKK/4 == 32 quads per key
            const int kq = (idx & 31) * 4;
            const float4 v = *(const float4*)(kp + (size_t)nl * DKK + kq);
            kls[nl][kq + 0] = (__bf16)v.x;
            kls[nl][kq + 1] = (__bf16)v.y;
            kls[nl][kq + 2] = (__bf16)v.z;
            kls[nl][kq + 3] = (__bf16)v.w;
            atomicAdd(&ssq[nl], v.x * v.x + v.y * v.y + v.z * v.z + v.w * v.w);
        }
        __syncthreads();
        for (int i = tid; i < TN; i += 256) rno[i] = w[s] / (sqrtf(ssq[i]) + EPS_);
        __syncthreads();

        // A fragments: 16x32 bf16 per k-chunk, rows = this wave's m-tile
        const __bf16* qrow = qnbf + ((size_t)s * B_B + (mtile * 16 + mrow)) * DKK;
        v16bf afrag[4];
#pragma unroll
        for (int kc = 0; kc < 4; ++kc) {
            const int k0 = kc * 32 + ((lane < 16) ? 0 : 8);
            const int k1 = kc * 32 + ((lane < 16) ? 16 : 24);
            const v8bf lo = *(const v8bf*)(qrow + k0);
            const v8bf hi = *(const v8bf*)(qrow + k1);
#pragma unroll
            for (int i = 0; i < 8; ++i) { afrag[kc][i] = lo[i]; afrag[kc][8 + i] = hi[i]; }
        }
#pragma unroll
        for (int t = 0; t < 4; ++t) {
            const int nsub = grp * 4 + t;
            const int ncol = nsub * 16 + (lane & 15);
            v8f tmp = {};
#pragma unroll
            for (int kc = 0; kc < 4; ++kc) {
                const int k0 = kc * 32 + ((lane < 16) ? 0 : 8);
                const int k1 = kc * 32 + ((lane < 16) ? 16 : 24);
                const v8bf lo = *(const v8bf*)(&kls[ncol][k0]);
                const v8bf hi = *(const v8bf*)(&kls[ncol][k1]);
                v16bf bfrag;
#pragma unroll
                for (int i = 0; i < 8; ++i) { bfrag[i] = lo[i]; bfrag[8 + i] = hi[i]; }
                tmp = __builtin_amdgcn_wmma_f32_16x16x32_bf16(
                        false, afrag[kc], false, bfrag, (short)0, tmp, false, false);
            }
            const float rs = rno[ncol];  // per-column (per-lane) w_s/(|k|+eps)
#pragma unroll
            for (int r = 0; r < 8; ++r) acc[t][r] += rs * tmp[r];
        }
        __syncthreads();   // LDS reused next aspect
    }
    // store C tiles: lane holds column lane&15, rows mtile*16 + {0..7|8..15}
#pragma unroll
    for (int t = 0; t < 4; ++t) {
        const int ncol  = nbase + (grp * 4 + t) * 16 + (lane & 15);
        const int rbase = mtile * 16 + ((lane < 16) ? 0 : 8);
#pragma unroll
        for (int r = 0; r < 8; ++r)
            sfull[(size_t)(rbase + r) * N_KEYS + ncol] = acc[t][r];
    }
}

// ===== K6: gather 4096 candidates, top-64, alphas + global_idx =======
__global__ __launch_bounds__(256) void k6_topk_cand(
        const float* __restrict__ sfull, const int* __restrict__ topc,
        const float* __restrict__ tau, const float* __restrict__ lam,
        const int* __restrict__ warm, float* __restrict__ out_alpha,
        float* __restrict__ out_idx) {
    __shared__ float vals[MCAND];   // 16 KB
    __shared__ float rv[256];
    __shared__ int   ri[256];
    __shared__ float sred[256];
    __shared__ float keep[K_MAX_];
    __shared__ int   kpos[K_MAX_];
    __shared__ float Rsum;
    const int b = blockIdx.x, tid = threadIdx.x;
    const float lm = lam[0], tv = tau[0];
    float racc = 0.f;
    for (int j = tid; j < MCAND; j += 256) {
        const int c = topc[b * M_TOP + (j >> 7)];
        const int n = c * NPC + (j & (NPC - 1));
        const float v = sfull[(size_t)b * N_KEYS + n];
        vals[j] = v;
        racc += sigmoidf_(lm * (v - tv)) * expf(v);
    }
    sred[tid] = racc;
    __syncthreads();
    for (int off = 128; off > 0; off >>= 1) {
        if (tid < off) sred[tid] += sred[tid + off];
        __syncthreads();
    }
    if (tid == 0) Rsum = sred[0];
    __syncthreads();
    for (int it = 0; it < K_MAX_; ++it) {
        float bv = NEG_INF; int bi = 0;
        for (int j = tid; j < MCAND; j += 256)
            if (vals[j] > bv) { bv = vals[j]; bi = j; }
        rv[tid] = bv; ri[tid] = bi;
        __syncthreads();
        for (int off = 128; off > 0; off >>= 1) {
            if (tid < off && rv[tid + off] > rv[tid]) { rv[tid] = rv[tid + off]; ri[tid] = ri[tid + off]; }
            __syncthreads();
        }
        if (tid == 0) { keep[it] = rv[0]; kpos[it] = ri[0]; vals[ri[0]] = NEG_INF; }
        __syncthreads();
    }
    if (tid == 0) {
        float a[K_MAX_];
        if (warm[0]) {
            const float m = keep[0];           // sorted descending
            float sum = 0.f;
            for (int i = 0; i < K_MAX_; ++i) { a[i] = expf(keep[i] - m); sum += a[i]; }
            for (int i = 0; i < K_MAX_; ++i) a[i] /= sum;
        } else {
            const float inv = 1.0f / (Rsum + EPS_);
            float sum = 0.f;
            for (int i = 0; i < K_MAX_; ++i) {
                a[i] = sigmoidf_(lm * (keep[i] - tv)) * expf(keep[i]) * inv;
                sum += a[i];
            }
            const float inv2 = 1.0f / (sum + EPS_);
            for (int i = 0; i < K_MAX_; ++i) a[i] *= inv2;
        }
        for (int i = 0; i < K_MAX_; ++i) {
            out_alpha[b * K_MAX_ + i] = a[i];
            const int p = kpos[i];
            const int c = topc[b * M_TOP + (p >> 7)];
            out_idx[b * K_MAX_ + i] = (float)(c * NPC + (p & (NPC - 1)));
        }
    }
}

// ============ K7: per-row max and sum-exp over N for softmax =========
__global__ __launch_bounds__(512) void k7_rowstats(
        const float* __restrict__ sfull, float* __restrict__ rmax,
        float* __restrict__ rsum) {
    __shared__ float red[512];
    __shared__ float mrow;
    const int b = blockIdx.x, tid = threadIdx.x;
    const float* row = sfull + (size_t)b * N_KEYS;
    float m = NEG_INF;
    for (int j = tid; j < N_KEYS; j += 512) m = fmaxf(m, row[j]);
    red[tid] = m;
    __syncthreads();
    for (int off = 256; off > 0; off >>= 1) {
        if (tid < off) red[tid] = fmaxf(red[tid], red[tid + off]);
        __syncthreads();
    }
    if (tid == 0) mrow = red[0];
    __syncthreads();
    float sm = 0.f;
    for (int j = tid; j < N_KEYS; j += 512) sm += expf(row[j] - mrow);
    red[tid] = sm;
    __syncthreads();
    for (int off = 256; off > 0; off >>= 1) {
        if (tid < off) red[tid] += red[tid + off];
        __syncthreads();
    }
    if (tid == 0) { rmax[b] = mrow; rsum[b] = red[0]; }
}

// ============ K8: in-place softmax of s_full -> soft_full ============
__global__ __launch_bounds__(256) void k8_softmax(
        float* __restrict__ sfull, const float* __restrict__ rmax,
        const float* __restrict__ rsum) {
    const size_t idx = (size_t)blockIdx.x * blockDim.x + threadIdx.x;
    if (idx >= (size_t)B_B * N_KEYS) return;
    const int b = (int)(idx >> 17);           // N_KEYS == 2^17
    sfull[idx] = expf(sfull[idx] - rmax[b]) / rsum[b];
}

// ============================ launcher ===============================
extern "C" void kernel_launch(void* const* d_in, const int* in_sizes, int n_in,
                              void* d_out, int out_size, void* d_ws, size_t ws_size,
                              hipStream_t stream) {
    const float* z    = (const float*)d_in[0];
    const float* pool = (const float*)d_in[1];
    const float* WQ   = (const float*)d_in[2];
    const float* aw   = (const float*)d_in[3];
    const float* tau  = (const float*)d_in[4];
    const float* cent = (const float*)d_in[5];
    const float* lam  = (const float*)d_in[6];
    const int*   warm = (const int*)d_in[7];

    char* ws = (char*)d_ws;
    float*  w_      = (float*)(ws + WS_W);
    float*  qnf     = (float*)(ws + WS_QNF);
    __bf16* qnbf    = (__bf16*)(ws + WS_QNBF);
    float*  cscore  = (float*)(ws + WS_CSCORE);
    int*    topc    = (int*)(ws + WS_TOPC);
    float*  rmax    = (float*)(ws + WS_RMAX);
    float*  rsum    = (float*)(ws + WS_RSUM);

    float* out = (float*)d_out;
    float* out_alpha = out;                                // [64,64]
    float* out_idx   = out + B_B * K_MAX_;                 // [64,64]
    float* sfull     = out + 2 * B_B * K_MAX_;             // [64,131072], raw then softmax in place

    k0_prep_w<<<1, 32, 0, stream>>>(aw, w_);
    k1_queries<<<B_B * S_ASP, 128, 0, stream>>>(z, WQ, qnf, qnbf);
    k2_centroids<<<B_B, 256, 0, stream>>>(qnf, cent, w_, cscore);
    k3_topclusters<<<B_B, 256, 0, stream>>>(cscore, topc);
    k4_dense<<<N_KEYS / TN, 256, 0, stream>>>(pool, w_, qnbf, sfull);
    k6_topk_cand<<<B_B, 256, 0, stream>>>(sfull, topc, tau, lam, warm, out_alpha, out_idx);
    k7_rowstats<<<B_B, 512, 0, stream>>>(sfull, rmax, rsum);
    const size_t total = (size_t)B_B * N_KEYS;
    k8_softmax<<<(int)((total + 255) / 256), 256, 0, stream>>>(sfull, rmax, rsum);
}
